// MultiHeadAttention_26302379721428
// MI455X (gfx1250) — compile-verified, matrix operand order checked by reading, not run on hardware
//
#include <hip/hip_runtime.h>
#include <hip/hip_bf16.h>
#include <stdint.h>

#define NHEAD 16
#define DMODEL 1024
#define BATCH 2
#define SEQ 2048
#define HDIM 64
#define MTOT (BATCH * SEQ) // 4096

typedef __bf16 bf16;
typedef bf16  v16bf __attribute__((ext_vector_type(16)));
typedef float v8f   __attribute__((ext_vector_type(8)));
typedef float v4f   __attribute__((ext_vector_type(4)));
typedef int   v4i   __attribute__((ext_vector_type(4)));
typedef int   v8i   __attribute__((ext_vector_type(8)));
typedef unsigned v4u __attribute__((ext_vector_type(4)));
typedef unsigned v2u __attribute__((ext_vector_type(2)));

static __device__ __forceinline__ unsigned short bf16_bits(float f) {
  unsigned u = __float_as_uint(f);
  u += 0x7FFFu + ((u >> 16) & 1u); // round-to-nearest-even
  return (unsigned short)(u >> 16);
}
static __device__ __forceinline__ bf16 f2bf(float f) {
  unsigned short h = bf16_bits(f);
  bf16 r;
  __builtin_memcpy(&r, &h, 2);
  return r;
}
static __device__ __forceinline__ v8f vzero8() {
  v8f v;
#pragma unroll
  for (int i = 0; i < 8; ++i) v[i] = 0.0f;
  return v;
}
// Load a 16-element bf16 WMMA fragment as two contiguous 16B chunks.
static __device__ __forceinline__ v16bf load_frag(const bf16* p0, const bf16* p1) {
  union { v4i u[2]; v16bf v; } f;
  f.u[0] = *(const v4i*)p0;
  f.u[1] = *(const v4i*)p1;
  return f.v;
}
static __device__ __forceinline__ void storeC(float* p, float x) { *p = x; }
static __device__ __forceinline__ void storeC(bf16* p, float x) { *p = f2bf(x); }

// Byte offset of a __shared__ object within the LDS segment (generic pointer
// to LDS keeps the LDS offset in the low 32 bits).
static __device__ __forceinline__ uint32_t lds_off(const void* p) {
  return (uint32_t)(uintptr_t)p;
}

// ---- Tensor Data Mover: 2-D bf16 tile load, global -> LDS (ISA ch.8 D#) ----
// tile_d0 = contiguous extent (elements), tile_d1 = rows, stride_d0 = row
// stride in elements. Tensor dims are set equal to the tile (tiles never OOB).
// clang-23 toolchain: 6-arg builtin (g0 v4u, g1 v8i, g2 v4i, g3 v4i, v8i, cpol).
static __device__ __forceinline__ void tdm_load_2d_bf16(
    uint32_t lds_addr, const void* gptr,
    uint32_t tile_d0, uint32_t tile_d1, uint32_t stride_d0) {
  uint64_t ga = (uint64_t)(uintptr_t)gptr;
  v4u g0;
  g0[0] = 1u;                                    // count=1, user descriptor
  g0[1] = lds_addr;                              // lds_addr [63:32]
  g0[2] = (uint32_t)ga;                          // global_addr[31:0]
  g0[3] = ((uint32_t)(ga >> 32) & 0x01FFFFFFu)   // global_addr[56:32]
          | (2u << 30);                          // type=2 ("image")
  v8i g1;
  g1[0] = (int)(1u << 16);                       // data_size=1 -> 2 bytes
  g1[1] = (int)((tile_d0 & 0xFFFFu) << 16);      // tensor_dim0[15:0] @ [63:48]
  g1[2] = (int)(((tile_d0 >> 16) & 0xFFFFu)      // tensor_dim0[31:16]
          | ((tile_d1 & 0xFFFFu) << 16));        // tensor_dim1[15:0]
  g1[3] = (int)(((tile_d1 >> 16) & 0xFFFFu)      // tensor_dim1[31:16]
          | ((tile_d0 & 0xFFFFu) << 16));        // tile_dim0 @ [127:112]
  g1[4] = (int)(tile_d1 & 0xFFFFu);              // tile_dim1 (tile_dim2 = 0)
  g1[5] = (int)stride_d0;                        // tensor_dim0_stride[31:0]
  g1[6] = 0;                                     // stride0[47:32], stride1 lo
  g1[7] = 0;                                     // stride1 hi (2-D: unused)
  v4i gz4 = {0, 0, 0, 0};                        // groups 2/3: unused (2-D)
  v8i gz8 = {0, 0, 0, 0, 0, 0, 0, 0};            // trailing group: unused (2-D)
  __builtin_amdgcn_tensor_load_to_lds(g0, g1, gz4, gz4, gz8, 0);
}

// ---------------- f32 -> bf16 conversion (vectorized x4) ----------------
__global__ __launch_bounds__(256) void cvt_f32_to_bf16(const float* __restrict__ in,
                                                       bf16* __restrict__ out, int n4) {
  int i = blockIdx.x * 256 + threadIdx.x;
  if (i < n4) {
    v4f v = *(const v4f*)(in + (size_t)i * 4);
    v2u pk;
    pk[0] = (unsigned)bf16_bits(v[0]) | ((unsigned)bf16_bits(v[1]) << 16);
    pk[1] = (unsigned)bf16_bits(v[2]) | ((unsigned)bf16_bits(v[3]) << 16);
    *(v2u*)(out + (size_t)i * 4) = pk;
  }
}

// ---------------- bf16 GEMM: C[M,N] = A[M,K] @ B[K,N] ----------------
// 256 threads = 8 waves; block tile 128x128, k-step 32; wave tile 32x64.
// A tiles are DMA'd by the TDM (double-buffered); B tiles are staged
// transposed [n][k] by the VALU (TDM cannot transpose).
template <typename OutT>
__global__ __launch_bounds__(256) void gemm_bf16_128x128(
    const bf16* __restrict__ A, const bf16* __restrict__ B, OutT* __restrict__ C,
    int M, int N, int K) {
  __shared__ __align__(16) bf16 sA[2][128 * 32]; // [m][k]
  __shared__ __align__(16) bf16 sB[2][128 * 32]; // [n][k] (transposed)
  const int tid = threadIdx.x;
  const int lane = tid & 31;
  const int wave = tid >> 5;
  const int wm = wave >> 1; // 0..3 -> 32 rows each
  const int wn = wave & 1;  // 0..1 -> 64 cols each
  const int m0 = blockIdx.y * 128;
  const int n0 = blockIdx.x * 128;
  const int lr = lane & 15;
  const int ko = (lane < 16) ? 0 : 8; // per-lane K sub-offset (ISA A/B layout)

  v8f acc[2][4];
#pragma unroll
  for (int i = 0; i < 2; ++i)
#pragma unroll
    for (int j = 0; j < 4; ++j) acc[i][j] = vzero8();

  const int bn = tid & 127, bk = (tid >> 7) * 16; // B loader: 16 k per thread

  auto issueA = [&](int buf, int kt) {
    if ((tid >> 5) == 0) // wave-uniform guard: exactly one wave issues the DMA
      tdm_load_2d_bf16(lds_off(&sA[buf][0]), A + (size_t)m0 * K + kt,
                       /*tile_d0=*/32, /*tile_d1=*/128, /*stride=*/(uint32_t)K);
  };
  auto stageB = [&](int buf, int kt) {
    __align__(16) bf16 tmp[16];
#pragma unroll
    for (int kk = 0; kk < 16; ++kk)
      tmp[kk] = B[(size_t)(kt + bk + kk) * N + n0 + bn];
    v4i* dst = (v4i*)(&sB[buf][bn * 32 + bk]);
    dst[0] = ((v4i*)tmp)[0];
    dst[1] = ((v4i*)tmp)[1];
  };

  issueA(0, 0);
  stageB(0, 0);
  __builtin_amdgcn_s_wait_tensorcnt(0);
  __syncthreads();

  const int nsteps = K / 32;
  for (int s = 0; s < nsteps; ++s) {
    const int cur = s & 1, nxt = cur ^ 1;
    if (s + 1 < nsteps) { // prefetch next tiles while computing current
      issueA(nxt, (s + 1) * 32);
      stageB(nxt, (s + 1) * 32);
    }
    v16bf af[2], bfr[4];
#pragma unroll
    for (int i = 0; i < 2; ++i) {
      const bf16* base = &sA[cur][(wm * 32 + i * 16 + lr) * 32 + ko];
      af[i] = load_frag(base, base + 16);
    }
#pragma unroll
    for (int j = 0; j < 4; ++j) {
      const bf16* base = &sB[cur][(wn * 64 + j * 16 + lr) * 32 + ko];
      bfr[j] = load_frag(base, base + 16);
    }
#pragma unroll
    for (int i = 0; i < 2; ++i)
#pragma unroll
      for (int j = 0; j < 4; ++j)
        acc[i][j] = __builtin_amdgcn_wmma_f32_16x16x32_bf16(
            false, af[i], false, bfr[j], (short)0, acc[i][j], false, false);
    __builtin_amdgcn_s_wait_tensorcnt(0); // wave0: next A tile landed
    __syncthreads();                      // everyone may flip buffers
  }
#pragma unroll
  for (int i = 0; i < 2; ++i)
#pragma unroll
    for (int j = 0; j < 4; ++j)
#pragma unroll
      for (int r = 0; r < 8; ++r) {
        int row = m0 + wm * 32 + i * 16 + r + ((lane < 16) ? 0 : 8);
        int col = n0 + wn * 64 + j * 16 + lr;
        storeC(&C[(size_t)row * N + col], acc[i][j][r]);
      }
}

// ---------------- flash attention: 1 WG = (b,h,64 q rows); 4 waves ----------------
__global__ __launch_bounds__(128) void flash_attn(
    const bf16* __restrict__ q, const bf16* __restrict__ k, const bf16* __restrict__ v,
    const int* __restrict__ mask, bf16* __restrict__ z) {
  __shared__ __align__(16) bf16 sK[64 * 64];      // [key][dim]   (TDM target)
  __shared__ __align__(16) bf16 sV[64 * 64];      // [dim][key]   (transposed)
  __shared__ __align__(16) bf16 sP[4][16 * 64];   // per-wave P tile [row][key]
  const int tid = threadIdx.x;
  const int lane = tid & 31;
  const int w = tid >> 5;
  const int lr = lane & 15;
  const int ko = (lane < 16) ? 0 : 8;
  const int b = blockIdx.y >> 4;
  const int h = blockIdx.y & 15;
  const int q0 = blockIdx.x * 64;
  const size_t base = (size_t)b * SEQ * DMODEL + (size_t)h * HDIM;

  // Q A-fragments (16 rows x 64 dims), held in registers for whole kernel
  v16bf qf[2];
  {
    const bf16* qp = q + base + (size_t)(q0 + w * 16 + lr) * DMODEL + ko;
    qf[0] = load_frag(qp, qp + 16);
    qf[1] = load_frag(qp + 32, qp + 48);
  }

  v8f O[4];
#pragma unroll
  for (int nt = 0; nt < 4; ++nt) O[nt] = vzero8();
  float Mr[8], Lr[8];
#pragma unroll
  for (int i = 0; i < 8; ++i) { Mr[i] = -__builtin_inff(); Lr[i] = 0.0f; }

  const int rbase = q0 + w * 16 + ((lane < 16) ? 0 : 8);

  for (int j = 0; j < SEQ / 64; ++j) {
    const int k0 = j * 64;
    __syncthreads(); // prev iteration's readers of sK/sV/sP are done
    // K tile via TDM: straight 64x64 strided copy, one wave issues
    if ((tid >> 5) == 0)
      tdm_load_2d_bf16(lds_off(sK), k + base + (size_t)k0 * DMODEL,
                       /*tile_d0=*/64, /*tile_d1=*/64, /*stride=*/DMODEL);
    { // V tile transposed [dim][key]: gather strided, write contiguous 64B
      int dim = tid & 63, kh = (tid >> 6) * 32;
      __align__(16) bf16 tmp[32];
#pragma unroll
      for (int kk = 0; kk < 32; ++kk)
        tmp[kk] = v[base + (size_t)(k0 + kh + kk) * DMODEL + dim];
      v4i* dst = (v4i*)(sV + dim * 64 + kh);
#pragma unroll
      for (int t4 = 0; t4 < 4; ++t4) dst[t4] = ((v4i*)tmp)[t4];
    }
    // prefetch next kv-block's mask rows into cache (global_prefetch_b8)
    if (k0 + 64 < SEQ) {
      const int* mp = mask + ((size_t)b * SEQ + rbase) * SEQ + (k0 + 64) + lr;
#pragma unroll
      for (int nt = 0; nt < 4; ++nt) __builtin_prefetch(mp + nt * 16, 0, 1);
    }
    __builtin_amdgcn_s_wait_tensorcnt(0); // wave0: K tile landed
    __syncthreads();

    // S = q @ k^T  (4 key-tiles of 16)
    v8f S[4];
#pragma unroll
    for (int nt = 0; nt < 4; ++nt) {
      S[nt] = vzero8();
#pragma unroll
      for (int kk = 0; kk < 2; ++kk) {
        const bf16* bp = sK + (nt * 16 + lr) * 64 + kk * 32 + ko;
        v16bf bfr = load_frag(bp, bp + 16);
        S[nt] = __builtin_amdgcn_wmma_f32_16x16x32_bf16(
            false, qf[kk], false, bfr, (short)0, S[nt], false, false);
      }
    }
    // scale + mask (C-layout: row = vgpr + 8*(lane>=16), col = lane%16)
#pragma unroll
    for (int nt = 0; nt < 4; ++nt)
#pragma unroll
      for (int i = 0; i < 8; ++i) {
        float s = S[nt][i] * 0.125f; // 1/sqrt(64)
        int mv = mask[((size_t)b * SEQ + (rbase + i)) * SEQ + k0 + nt * 16 + lr];
        S[nt][i] = mv ? s : -1.0e9f;
      }
    // online softmax per query row (reduce across 16-lane half via shfl_xor)
#pragma unroll
    for (int i = 0; i < 8; ++i) {
      float rm = fmaxf(fmaxf(S[0][i], S[1][i]), fmaxf(S[2][i], S[3][i]));
      rm = fmaxf(rm, __shfl_xor(rm, 1, 32));
      rm = fmaxf(rm, __shfl_xor(rm, 2, 32));
      rm = fmaxf(rm, __shfl_xor(rm, 4, 32));
      rm = fmaxf(rm, __shfl_xor(rm, 8, 32));
      float newM = fmaxf(Mr[i], rm);
      float alpha = __expf(Mr[i] - newM);
      float rs = 0.0f;
#pragma unroll
      for (int nt = 0; nt < 4; ++nt) {
        float p = __expf(S[nt][i] - newM);
        S[nt][i] = p;
        rs += p;
      }
      rs += __shfl_xor(rs, 1, 32);
      rs += __shfl_xor(rs, 2, 32);
      rs += __shfl_xor(rs, 4, 32);
      rs += __shfl_xor(rs, 8, 32);
      Lr[i] = Lr[i] * alpha + rs;
      Mr[i] = newM;
#pragma unroll
      for (int nt = 0; nt < 4; ++nt) O[nt][i] *= alpha;
    }
    // re-layout P (C-layout -> A-layout) via LDS
    {
      bf16* p = sP[w];
      int roff = (lane < 16) ? 0 : 8;
#pragma unroll
      for (int nt = 0; nt < 4; ++nt)
#pragma unroll
        for (int i = 0; i < 8; ++i)
          p[(i + roff) * 64 + nt * 16 + lr] = f2bf(S[nt][i]);
    }
    __syncthreads();
    // O += P @ V
    v16bf pf[2];
    {
      const bf16* pb = sP[w] + lr * 64 + ko;
      pf[0] = load_frag(pb, pb + 16);
      pf[1] = load_frag(pb + 32, pb + 48);
    }
#pragma unroll
    for (int nt = 0; nt < 4; ++nt)
#pragma unroll
      for (int kk = 0; kk < 2; ++kk) {
        const bf16* vp = sV + (nt * 16 + lr) * 64 + kk * 32 + ko;
        v16bf vf = load_frag(vp, vp + 16);
        O[nt] = __builtin_amdgcn_wmma_f32_16x16x32_bf16(
            false, pf[kk], false, vf, (short)0, O[nt], false, false);
      }
  }
  // normalize + store z (bf16, feeds final GEMM)
#pragma unroll
  for (int nt = 0; nt < 4; ++nt)
#pragma unroll
    for (int i = 0; i < 8; ++i) {
      float val = O[nt][i] / Lr[i];
      z[((size_t)b * SEQ + rbase + i) * DMODEL + h * HDIM + nt * 16 + lr] = f2bf(val);
    }
}

extern "C" void kernel_launch(void* const* d_in, const int* in_sizes, int n_in,
                              void* d_out, int out_size, void* d_ws, size_t ws_size,
                              hipStream_t stream) {
  const float* Q   = (const float*)d_in[0];
  const float* K   = (const float*)d_in[1];
  const float* V   = (const float*)d_in[2];
  const int*  mask = (const int*)d_in[3];
  const float* Wq  = (const float*)d_in[4];
  const float* Wk  = (const float*)d_in[5];
  const float* Wv  = (const float*)d_in[6];
  const float* Wo  = (const float*)d_in[7];
  float* out = (float*)d_out;

  // workspace layout (64 MB total, bf16 staging)
  char* ws = (char*)d_ws;
  size_t off = 0;
  auto take = [&](size_t bytes) {
    char* p = ws + off;
    off += (bytes + 255) & ~(size_t)255;
    return p;
  };
  const size_t actB = (size_t)MTOT * DMODEL * sizeof(bf16);   // 8 MB
  const size_t wB   = (size_t)DMODEL * DMODEL * sizeof(bf16); // 2 MB
  bf16* Qb  = (bf16*)take(actB);
  bf16* Kb  = (bf16*)take(actB);
  bf16* Vb  = (bf16*)take(actB);
  bf16* Wqb = (bf16*)take(wB);
  bf16* Wkb = (bf16*)take(wB);
  bf16* Wvb = (bf16*)take(wB);
  bf16* Wob = (bf16*)take(wB);
  bf16* qp  = (bf16*)take(actB);
  bf16* kp  = (bf16*)take(actB);
  bf16* vp  = (bf16*)take(actB);
  bf16* zb  = (bf16*)take(actB);

  const int n4a = MTOT * DMODEL / 4;
  const int n4w = DMODEL * DMODEL / 4;
  cvt_f32_to_bf16<<<(n4a + 255) / 256, 256, 0, stream>>>(Q, Qb, n4a);
  cvt_f32_to_bf16<<<(n4a + 255) / 256, 256, 0, stream>>>(K, Kb, n4a);
  cvt_f32_to_bf16<<<(n4a + 255) / 256, 256, 0, stream>>>(V, Vb, n4a);
  cvt_f32_to_bf16<<<(n4w + 255) / 256, 256, 0, stream>>>(Wq, Wqb, n4w);
  cvt_f32_to_bf16<<<(n4w + 255) / 256, 256, 0, stream>>>(Wk, Wkb, n4w);
  cvt_f32_to_bf16<<<(n4w + 255) / 256, 256, 0, stream>>>(Wv, Wvb, n4w);
  cvt_f32_to_bf16<<<(n4w + 255) / 256, 256, 0, stream>>>(Wo, Wob, n4w);

  dim3 gg(DMODEL / 128, MTOT / 128); // (8, 32)
  gemm_bf16_128x128<bf16><<<gg, 256, 0, stream>>>(Qb, Wqb, qp, MTOT, DMODEL, DMODEL);
  gemm_bf16_128x128<bf16><<<gg, 256, 0, stream>>>(Kb, Wkb, kp, MTOT, DMODEL, DMODEL);
  gemm_bf16_128x128<bf16><<<gg, 256, 0, stream>>>(Vb, Wvb, vp, MTOT, DMODEL, DMODEL);

  dim3 fg(SEQ / 64, BATCH * NHEAD); // (32, 32)
  flash_attn<<<fg, 128, 0, stream>>>(qp, kp, vp, mask, zb);

  gemm_bf16_128x128<float><<<gg, 256, 0, stream>>>(zb, Wob, out, MTOT, DMODEL, DMODEL);
}
